// SelfAttention_53807350284716
// MI455X (gfx1250) — matrix operand order repaired and checked by reading, not compiled, not run on hardware
//
#include <hip/hip_runtime.h>
#include <hip/hip_bf16.h>

typedef __attribute__((ext_vector_type(16))) _Float16 v16h;
typedef __attribute__((ext_vector_type(8)))  _Float16 v8h;
typedef __attribute__((ext_vector_type(4)))  _Float16 v4h;
typedef __attribute__((ext_vector_type(8)))  float    v8f;

#define SEQ 2048
#define NB  4
#define DIM 1024   // M = DQ = DV

union AB { v16h v16; v8h v8[2]; };

static __device__ __forceinline__ v8f vzero8() {
  v8f z = {0.f, 0.f, 0.f, 0.f, 0.f, 0.f, 0.f, 0.f};
  return z;
}

static __device__ __forceinline__ v8f wmma_f16(const AB& a, const AB& b, v8f c) {
  // D = A(16x32 f16) * B(32x16 f16) + C(16x16 f32)
  return __builtin_amdgcn_wmma_f32_16x16x32_f16(false, a.v16, false, b.v16,
                                                (short)0, c, false, false);
}

// low 32 bits of a generic pointer to LDS = LDS byte offset (aperture in [63:32])
static __device__ __forceinline__ unsigned lds_off(const void* p) {
  return (unsigned)(size_t)p;
}

static __device__ __forceinline__ void async_copy_b128(unsigned lds_addr,
                                                       const void* gaddr) {
  asm volatile("global_load_async_to_lds_b128 %0, %1, off"
               :: "v"(lds_addr), "v"(gaddr) : "memory");
}

static __device__ __forceinline__ void wait_asynccnt0() {
#if __has_builtin(__builtin_amdgcn_s_wait_asynccnt)
  __builtin_amdgcn_s_wait_asynccnt(0);
#else
  asm volatile("s_wait_asynccnt 0" ::: "memory");
#endif
}

// ---------------------------------------------------------------------------
// Kernel 1: fused QKV projection.
//   out = (X @ W^T + b) [* 1/sqrt(DQ) for Q]
//   Q, K stored row-major (n, k, d) f16; V stored transposed (n, d, k) f16.
// Grid: (DIM/64, SEQ/128, NB*3). Block: 256 (8 waves).
// ---------------------------------------------------------------------------
__global__ __launch_bounds__(256)
void qkv_proj_kernel(const float* __restrict__ query, const float* __restrict__ key,
                     const float* __restrict__ value,
                     const float* __restrict__ Wq, const float* __restrict__ bq,
                     const float* __restrict__ Wk, const float* __restrict__ bk,
                     const float* __restrict__ Wv, const float* __restrict__ bv,
                     _Float16* __restrict__ Qf, _Float16* __restrict__ Kf,
                     _Float16* __restrict__ Vt)
{
  constexpr int AP = 40;                 // padded LDS row stride in halves (80 B)
  __shared__ _Float16 As[128 * AP];      // 128 rows x 32 m  (input tile)
  __shared__ _Float16 Bs[64 * AP];       // 64  rows x 32 m  (weight tile, rows over d)

  const int tid = threadIdx.x;
  const int z = blockIdx.z;
  const int n = z / 3, p = z % 3;
  const int d0 = blockIdx.x * 64;
  const int k0 = blockIdx.y * 128;

  const float* X = (p == 0) ? query : (p == 1) ? key : value;
  const float* W = (p == 0) ? Wq    : (p == 1) ? Wk  : Wv;
  const float* Bv = (p == 0) ? bq   : (p == 1) ? bk  : bv;
  X += (size_t)n * SEQ * DIM;

  const int w     = tid >> 5;
  const int lane  = tid & 31;
  const int mr    = lane & 15;   // row (A) / col (B,D) within 16
  const int hi    = lane >> 4;
  const int base0 = hi * 8;      // unused; A-frag also loads at kbase
  const int kbase = hi * 16;     // B-frag K offset
  const int qr    = w * 16;      // wave's 16-row slice of the 128-row tile

  v8f acc[4];
  for (int t = 0; t < 4; ++t) acc[t] = vzero8();

  for (int m0 = 0; m0 < DIM; m0 += 32) {
    // stage input tile: 128x32 f32 -> f16
    for (int i = 0; i < 4; ++i) {
      int u = tid + i * 256;           // float4 index, 1024 total
      int r = u >> 3, c4 = u & 7;
      const float4 f = *(const float4*)(X + (size_t)(k0 + r) * DIM + m0 + c4 * 4);
      v4h h; h.x = (_Float16)f.x; h.y = (_Float16)f.y;
      h.z = (_Float16)f.z; h.w = (_Float16)f.w;
      *(v4h*)&As[r * AP + c4 * 4] = h;
    }
    // stage weight tile: 64x32 f32 -> f16 (rows over d; already B^T layout)
    for (int i = 0; i < 2; ++i) {
      int u = tid + i * 256;           // 512 float4 total
      int r = u >> 3, c4 = u & 7;
      const float4 f = *(const float4*)(W + (size_t)(d0 + r) * DIM + m0 + c4 * 4);
      v4h h; h.x = (_Float16)f.x; h.y = (_Float16)f.y;
      h.z = (_Float16)f.z; h.w = (_Float16)f.w;
      *(v4h*)&Bs[r * AP + c4 * 4] = h;
    }
    __syncthreads();

    // preload A and all 4 B fragments, then issue WMMAs back-to-back
    AB a, b[4];
    a.v8[0] = *(const v8h*)&As[(qr + mr) * AP + kbase];
    a.v8[1] = *(const v8h*)&As[(qr + mr) * AP + kbase + 8];
    for (int t = 0; t < 4; ++t) {
      const _Float16* bp = &Bs[(t * 16 + mr) * AP + kbase];
      b[t].v8[0] = *(const v8h*)bp;
      b[t].v8[1] = *(const v8h*)(bp + 8);
    }
    for (int t = 0; t < 4; ++t)
      acc[t] = wmma_f16(a, b[t], acc[t]);
    __syncthreads();
  }

  const float scale = (p == 0) ? 0.03125f : 1.0f;   // 1/sqrt(1024) folded into Q
  if (p < 2) {
    _Float16* out = ((p == 0) ? Qf : Kf) + (size_t)n * SEQ * DIM;
    for (int t = 0; t < 4; ++t) {
      const int d = d0 + t * 16 + mr;
      const float bias = Bv[d];
      const int krow = k0 + qr + hi * 8;
      for (int r = 0; r < 8; ++r)
        out[(size_t)(krow + r) * DIM + d] = (_Float16)((acc[t][r] + bias) * scale);
    }
  } else {
    // transposed store: lane owns column d, 8 consecutive k rows -> one b128
    for (int t = 0; t < 4; ++t) {
      const int d = d0 + t * 16 + mr;
      const float bias = Bv[d];
      v8h hv;
      for (int r = 0; r < 8; ++r) hv[r] = (_Float16)(acc[t][r] + bias);
      *(v8h*)(Vt + ((size_t)n * DIM + d) * SEQ + k0 + qr + hi * 8) = hv;
    }
  }
}

// ---------------------------------------------------------------------------
// Kernel 2: causal flash attention. One block = (batch n, 32 query rows).
// Q tile async-copied to LDS; K,V fragments streamed from L2-resident buffers.
// Grid: (SEQ/32, NB). Block: 256 (8 waves: 2 q-groups x 4 k/d-groups).
// ---------------------------------------------------------------------------
__global__ __launch_bounds__(256)
void flash_attn_kernel(const _Float16* __restrict__ Qf, const _Float16* __restrict__ Kf,
                       const _Float16* __restrict__ Vt, float* __restrict__ out)
{
  constexpr int QP = 1032;               // Q tile row stride (halves), 16B aligned
  constexpr int SP = 68;                 // S row stride (floats)
  constexpr int PP = 72;                 // P row stride (halves), 16B aligned
  __shared__ _Float16 Qs[32 * QP];
  __shared__ float    Ssm[32 * SP];
  __shared__ _Float16 Ps[32 * PP];
  __shared__ float    mrow[32], lrow[32], crow[32];

  const int tid   = threadIdx.x;
  const int lane  = tid & 31;
  const int w     = tid >> 5;
  const int mr    = lane & 15;
  const int hi    = lane >> 4;
  const int base0 = hi * 8;
  const int kbase = hi * 16;

  const int qb = blockIdx.x, n = blockIdx.y;
  const int q0 = qb * 32;

  const _Float16* Qg = Qf + ((size_t)n * SEQ + q0) * DIM;
  const _Float16* Kg = Kf + (size_t)n * SEQ * DIM;
  const _Float16* Vg = Vt + (size_t)n * DIM * SEQ;

  // async-stage Q tile (32 x 1024 f16) into LDS, bypassing VGPRs
  for (int i = 0; i < 16; ++i) {
    int u = tid + i * 256;               // v8h index, 4096 total
    int r = u >> 7, c8 = u & 127;
    async_copy_b128(lds_off(&Qs[r * QP + c8 * 8]),
                    (const void*)(Qg + (size_t)r * DIM + c8 * 8));
  }
  if (tid < 32) { mrow[tid] = -3.0e38f; lrow[tid] = 0.f; }

  const int qg = w & 1;                  // 16-row group for S and O
  const int dg = w >> 1;                 // 4 groups: 16 k-cols (S) / 256 d-cols (O)

  v8f o[16];
  for (int t = 0; t < 16; ++t) o[t] = vzero8();

  wait_asynccnt0();
  __syncthreads();

  const int jmax = (q0 + 31) >> 6;       // causal limit in 64-key blocks
  for (int j = 0; j <= jmax; ++j) {
    // prefetch next K block (L2-resident; speculative)
    if (j < jmax)
      __builtin_prefetch(Kg + (size_t)((j + 1) * 64 + dg * 16 + mr) * DIM, 0, 1);

    // ---- S = (Q/sqrt(d)) K^T : each wave one 16x16 fragment of 32x64 ----
    v8f s = vzero8();
    const int arow = (qg * 16 + mr) * QP;
    const _Float16* kp = Kg + (size_t)(j * 64 + dg * 16 + mr) * DIM + kbase;
    for (int c0 = 0; c0 < DIM; c0 += 32) {
      AB a, b;
      a.v8[0] = *(const v8h*)&Qs[arow + c0 + kbase];
      a.v8[1] = *(const v8h*)&Qs[arow + c0 + kbase + 8];
      b.v8[0] = *(const v8h*)(kp + c0);
      b.v8[1] = *(const v8h*)(kp + c0 + 8);
      s = wmma_f16(a, b, s);
    }
    // causal mask + spill S fragment to LDS
    {
      const int qrow = q0 + qg * 16 + hi * 8;
      const int kcol = j * 64 + dg * 16 + mr;
      const int srow = qg * 16 + hi * 8;
      for (int r = 0; r < 8; ++r) {
        float v = s[r];
        if (kcol > qrow + r) v = -1.0e9f;
        Ssm[(srow + r) * SP + dg * 16 + mr] = v;
      }
    }
    __syncthreads();

    // ---- online softmax: one thread per query row ----
    if (tid < 32) {
      const float* sr = &Ssm[tid * SP];
      float mold = mrow[tid];
      float mmax = mold;
      for (int c = 0; c < 64; ++c) mmax = fmaxf(mmax, sr[c]);
      const float corr = __expf(mold - mmax);
      float sum = 0.f;
      _Float16* pr = &Ps[tid * PP];
      for (int c = 0; c < 64; ++c) {
        float e = __expf(sr[c] - mmax);
        sum += e;
        pr[c] = (_Float16)e;
      }
      lrow[tid] = lrow[tid] * corr + sum;
      mrow[tid] = mmax;
      crow[tid] = corr;
    }
    __syncthreads();

    // ---- O = O*corr + P V : wave owns 16 q-rows x 256 d-cols ----
    {
      v8f cv;
      const int rbase = qg * 16 + hi * 8;
      for (int r = 0; r < 8; ++r) cv[r] = crow[rbase + r];
      for (int t = 0; t < 16; ++t) o[t] *= cv;

      const int prow = (qg * 16 + mr) * PP;
      for (int ks = 0; ks < 2; ++ks) {
        AB a;
        a.v8[0] = *(const v8h*)&Ps[prow + ks * 32 + kbase];
        a.v8[1] = *(const v8h*)&Ps[prow + ks * 32 + kbase + 8];
        const _Float16* vp = Vg + (size_t)(dg * 256 + mr) * SEQ + j * 64 + ks * 32 + kbase;
        for (int t = 0; t < 16; ++t) {
          AB b;
          const _Float16* vt = vp + (size_t)t * 16 * SEQ;
          b.v8[0] = *(const v8h*)vt;
          b.v8[1] = *(const v8h*)(vt + 8);
          o[t] = wmma_f16(a, b, o[t]);
        }
      }
    }
    __syncthreads();
  }

  // ---- normalize by 1/l and store fp32 ----
  {
    const int rbase = qg * 16 + hi * 8;
    v8f linv;
    for (int r = 0; r < 8; ++r) linv[r] = 1.0f / lrow[rbase + r];
    float* og = out + ((size_t)n * SEQ + q0) * DIM;
    for (int t = 0; t < 16; ++t) {
      const int d = dg * 256 + t * 16 + mr;
      for (int r = 0; r < 8; ++r)
        og[(size_t)(rbase + r) * DIM + d] = o[t][r] * linv[r];
    }
  }
}

extern "C" void kernel_launch(void* const* d_in, const int* in_sizes, int n_in,
                              void* d_out, int out_size, void* d_ws, size_t ws_size,
                              hipStream_t stream) {
  (void)in_sizes; (void)n_in; (void)out_size; (void)ws_size;
  const float* query = (const float*)d_in[0];
  const float* key   = (const float*)d_in[1];
  const float* value = (const float*)d_in[2];
  // d_in[3] is the causal mask; applied analytically in-kernel.
  const float* Wq = (const float*)d_in[4];
  const float* bq = (const float*)d_in[5];
  const float* Wk = (const float*)d_in[6];
  const float* bk = (const float*)d_in[7];
  const float* Wv = (const float*)d_in[8];
  const float* bv = (const float*)d_in[9];

  _Float16* ws = (_Float16*)d_ws;
  const size_t QSZ = (size_t)NB * SEQ * DIM;
  _Float16* Qf = ws;
  _Float16* Kf = ws + QSZ;
  _Float16* Vt = ws + 2 * QSZ;

  qkv_proj_kernel<<<dim3(DIM / 64, SEQ / 128, NB * 3), 256, 0, stream>>>(
      query, key, value, Wq, bq, Wk, bk, Wv, bv, Qf, Kf, Vt);
  flash_attn_kernel<<<dim3(SEQ / 32, NB), 256, 0, stream>>>(
      Qf, Kf, Vt, (float*)d_out);
}
